// CrossViewAttention_77876347011671
// MI455X (gfx1250) — compile-verified, matrix operand order checked by reading, not run on hardware
//
#include <hip/hip_runtime.h>
#include <cstdint>
#include <cstddef>

// Problem constants (B=1)
#define MQ    4096      // S*S query tokens; also H*W distinct context rows
#define CCH   256       // channels
#define NMLP  512       // MLP hidden
#define WIMG  128       // grd_x width

// Set to 0 if the assembler rejects the async-to-LDS mnemonics.
#define USE_ASYNC_COPY 1

typedef __attribute__((ext_vector_type(16))) _Float16 v16h;
typedef __attribute__((ext_vector_type(2)))  _Float16 h2;
typedef __attribute__((ext_vector_type(8)))  float    v8f;

// ---------------------------------------------------------------------------
// (C, M) -> (M, C) transpose (grd2sat -> x0, grd_x -> G)
// ---------------------------------------------------------------------------
__global__ __launch_bounds__(256)
void transpose_cm_kernel(const float* __restrict__ src, float* __restrict__ dst,
                         int M, int C) {
    int idx = blockIdx.x * blockDim.x + threadIdx.x;
    if (idx >= M * C) return;
    int m = idx % M;           // consecutive threads -> consecutive m: coalesced read
    int c = idx / M;
    dst[(size_t)m * C + c] = src[idx];
}

// ---------------------------------------------------------------------------
// gather indices: wl = clip(floor(u),0,W-1), wr = clip(floor(u)+1,0,W-1)
// ---------------------------------------------------------------------------
__global__ __launch_bounds__(256)
void make_idx_kernel(const float* __restrict__ u, int* __restrict__ wl,
                     int* __restrict__ wr, int n, int wmax) {
    int i = blockIdx.x * blockDim.x + threadIdx.x;
    if (i >= n) return;
    float uf = floorf(u[i]);
    wl[i] = (int)fminf(fmaxf(uf, 0.f), (float)wmax);
    wr[i] = (int)fminf(fmaxf(uf + 1.f, 0.f), (float)wmax);
}

// ---------------------------------------------------------------------------
// Weight convert fp32 -> fp16 WITH transpose: src is (K,N) row-major,
// dst is (N,K) row-major. B tiles then load exactly like A tiles.
// ---------------------------------------------------------------------------
__global__ __launch_bounds__(256)
void w_to_f16t_kernel(const float* __restrict__ src, _Float16* __restrict__ dst,
                      int K, int N) {
    int idx = blockIdx.x * blockDim.x + threadIdx.x;
    if (idx >= K * N) return;
    int n = idx % N;           // coalesced read
    int k = idx / N;
    dst[(size_t)n * K + k] = (_Float16)src[idx];
}

// ---------------------------------------------------------------------------
// Row LayerNorm: one wave32 per 256-wide row; writes f16 and/or f32 outputs.
// ---------------------------------------------------------------------------
__global__ __launch_bounds__(256)
void ln_rows_kernel(const float* __restrict__ X, const float* __restrict__ w,
                    const float* __restrict__ b, _Float16* __restrict__ o16,
                    float* __restrict__ o32, int C) {
    int wid  = threadIdx.x >> 5;
    int lane = threadIdx.x & 31;
    int row  = blockIdx.x * 8 + wid;
    const float* xr = X + (size_t)row * C;
    float vals[8];
    float s = 0.f, ss = 0.f;
    #pragma unroll
    for (int j = 0; j < 8; ++j) {
        float v = xr[lane + j * 32];
        vals[j] = v; s += v; ss += v * v;
    }
    #pragma unroll
    for (int off = 16; off > 0; off >>= 1) {
        s  += __shfl_xor(s,  off, 32);
        ss += __shfl_xor(ss, off, 32);
    }
    float mu  = s / (float)C;
    float var = ss / (float)C - mu * mu;
    float rs  = rsqrtf(var + 1e-5f);
    #pragma unroll
    for (int j = 0; j < 8; ++j) {
        int c = lane + j * 32;
        float y = (vals[j] - mu) * rs * w[c] + b[c];
        if (o16) o16[(size_t)row * C + c] = (_Float16)y;
        if (o32) o32[(size_t)row * C + c] = y;
    }
}

// ---------------------------------------------------------------------------
// WMMA GEMM: C[M,N] = A[M,K](f16, row-major) x Bt[N,K](f16, row-major = B^T).
// 256 threads / 8 waves; 128x64 tile, BK=32, double-buffered LDS with
// CDNA5 async global->LDS copies. Each wave: 32x32 output = 4 WMMA / K-step.
// ---------------------------------------------------------------------------
#define BM 128
#define BN 64
#define BK 32
#define LSTR 40   // padded LDS row stride in halfs (conflict-free, 16B-aligned)

__device__ inline void copy16_g2l(_Float16* dst_lds, const _Float16* src_g) {
#if USE_ASYNC_COPY
    // Per-lane 16B global->LDS DMA, tracked by ASYNCcnt.
    asm volatile("global_load_async_to_lds_b128 %0, %1, off"
                 :: "v"((uint32_t)(uintptr_t)dst_lds), "v"(src_g)
                 : "memory");
#else
    *reinterpret_cast<uint4*>(dst_lds) =
        *reinterpret_cast<const uint4*>(src_g);
#endif
}
__device__ inline void wait_async_prefetched() {
#if USE_ASYNC_COPY
    asm volatile("s_wait_asynccnt 0x3" ::: "memory");   // keep next tile in flight
#endif
}
__device__ inline void wait_async_all() {
#if USE_ASYNC_COPY
    asm volatile("s_wait_asynccnt 0x0" ::: "memory");
#endif
}

__device__ inline v16h load_frag(const _Float16* row, int lhi) {
    // 16-bit A-layout fragment: element e=2j(+1) -> K = (j<4?0:16)+lhi*8+(j%4)*2
    v16h f;
    #pragma unroll
    for (int j = 0; j < 8; ++j) {
        int kk = ((j & 4) << 2) + lhi * 8 + (j & 3) * 2;
        h2 t = *reinterpret_cast<const h2*>(row + kk);
        f[2 * j] = t[0]; f[2 * j + 1] = t[1];
    }
    return f;
}

template <bool BIAS, bool GELU, bool RESID, bool OUT16>
__global__ __launch_bounds__(256)
void gemm_f16_wmma_kernel(const _Float16* __restrict__ A,
                          const _Float16* __restrict__ Bt,
                          float* __restrict__ Cf, _Float16* __restrict__ Ch,
                          const float* __restrict__ bias,
                          const float* __restrict__ resid,
                          int M, int N, int K) {
    __shared__ _Float16 As[2][BM * LSTR];   // 2 x 10240 B
    __shared__ _Float16 Bs[2][BN * LSTR];   // 2 x  5120 B

    const int tid  = threadIdx.x;
    const int lane = tid & 31;
    const int wid  = tid >> 5;     // 0..7
    const int wm   = wid & 3;      // 32-row sub-tile
    const int wn   = wid >> 2;     // 0..1 -> 32-col sub-tile
    const int m0   = blockIdx.x * BM;
    const int n0   = blockIdx.y * BN;
    const int lhi  = (lane >> 4) & 1;
    const int lm   = lane & 15;

    // tile-load mapping: 4 threads per row, 8 halfs (16B) per thread
    const int ar = tid >> 2;             // 0..63 (A handles ar and ar+64)
    const int ac = (tid & 3) * 8;

    auto load_tile = [&](int kt, int buf) {
        const int k0 = kt * BK;
        copy16_g2l(&As[buf][ar * LSTR + ac],
                   A + (size_t)(m0 + ar) * K + (k0 + ac));
        copy16_g2l(&As[buf][(ar + 64) * LSTR + ac],
                   A + (size_t)(m0 + ar + 64) * K + (k0 + ac));
        copy16_g2l(&Bs[buf][ar * LSTR + ac],
                   Bt + (size_t)(n0 + ar) * K + (k0 + ac));
    };

    v8f acc[2][2] = {};

    const int nk = K / BK;
    load_tile(0, 0);
    for (int kt = 0; kt < nk; ++kt) {
        const int cur = kt & 1;
        if (kt + 1 < nk) {
            load_tile(kt + 1, cur ^ 1);
            wait_async_prefetched();
        } else {
            wait_async_all();
        }
        __syncthreads();

        v16h a0 = load_frag(&As[cur][(wm * 32 + lm) * LSTR], lhi);
        v16h a1 = load_frag(&As[cur][(wm * 32 + 16 + lm) * LSTR], lhi);
        v16h b0 = load_frag(&Bs[cur][(wn * 32 + lm) * LSTR], lhi);
        v16h b1 = load_frag(&Bs[cur][(wn * 32 + 16 + lm) * LSTR], lhi);

        acc[0][0] = __builtin_amdgcn_wmma_f32_16x16x32_f16(false, a0, false, b0,
                                                           (short)0, acc[0][0], false, false);
        acc[0][1] = __builtin_amdgcn_wmma_f32_16x16x32_f16(false, a0, false, b1,
                                                           (short)0, acc[0][1], false, false);
        acc[1][0] = __builtin_amdgcn_wmma_f32_16x16x32_f16(false, a1, false, b0,
                                                           (short)0, acc[1][0], false, false);
        acc[1][1] = __builtin_amdgcn_wmma_f32_16x16x32_f16(false, a1, false, b1,
                                                           (short)0, acc[1][1], false, false);
        __syncthreads();
    }

    // Epilogue. D layout: vgpr v -> m_local = v + 8*lhi ; n_local = lane&15
    #pragma unroll
    for (int ai = 0; ai < 2; ++ai) {
        #pragma unroll
        for (int v = 0; v < 8; ++v) {
            int mrow = m0 + wm * 32 + ai * 16 + v + lhi * 8;
            #pragma unroll
            for (int bi = 0; bi < 2; ++bi) {
                int ncol = n0 + wn * 32 + bi * 16 + lm;
                float val = acc[ai][bi][v];
                if (BIAS)  val += bias[ncol];
                if (GELU)  val = 0.5f * val * (1.0f + erff(val * 0.70710678118654752f));
                if (RESID) val += resid[(size_t)mrow * N + ncol];
                if (OUT16) Ch[(size_t)mrow * N + ncol] = (_Float16)val;
                else       Cf[(size_t)mrow * N + ncol] = val;
            }
        }
    }
}

// ---------------------------------------------------------------------------
// Cross attention over 64 gathered context rows. Thread = (query, head).
// K/V rows indexed r = h*128 + w  (n<32 -> w_left, n>=32 -> w_right).
// ---------------------------------------------------------------------------
__global__ __launch_bounds__(128)
void attn_kernel(const float* __restrict__ Q, const float* __restrict__ Km,
                 const float* __restrict__ Vm, const int* __restrict__ wl,
                 const int* __restrict__ wr, _Float16* __restrict__ Ao) {
    __shared__ float sc[128 * 64];   // 32 KB scores
    const int t    = threadIdx.x;
    const int mloc = t >> 3;         // 16 queries / block
    const int head = t & 7;
    const int m    = blockIdx.x * 16 + mloc;
    const int wlv  = wl[m];
    const int wrv  = wr[m];

    float qr[32];
    const float* qp = Q + (size_t)m * CCH + head * 32;
    #pragma unroll
    for (int d = 0; d < 32; ++d) qr[d] = qp[d];

    float* srow = &sc[t * 64];
    float mx = -1e30f;
    for (int n = 0; n < 64; ++n) {
        int row = (n < 32) ? (n * WIMG + wlv) : ((n - 32) * WIMG + wrv);
        const float* kp = Km + (size_t)row * CCH + head * 32;
        float dot = 0.f;
        #pragma unroll
        for (int d = 0; d < 32; ++d) dot += qr[d] * kp[d];
        dot *= 0.17677669529663687f;   // 1/sqrt(32)
        srow[n] = dot;
        mx = fmaxf(mx, dot);
    }
    float se = 0.f;
    for (int n = 0; n < 64; ++n) {
        float p = __expf(srow[n] - mx);
        srow[n] = p;
        se += p;
    }
    float inv = 1.f / se;

    float acc[32];
    #pragma unroll
    for (int d = 0; d < 32; ++d) acc[d] = 0.f;
    for (int n = 0; n < 64; ++n) {
        int row = (n < 32) ? (n * WIMG + wlv) : ((n - 32) * WIMG + wrv);
        const float* vp = Vm + (size_t)row * CCH + head * 32;
        float p = srow[n];
        #pragma unroll
        for (int d = 0; d < 32; ++d) acc[d] += p * vp[d];
    }
    _Float16* out = Ao + (size_t)m * CCH + head * 32;
    #pragma unroll
    for (int d = 0; d < 32; ++d) out[d] = (_Float16)(acc[d] * inv);
}

// ---------------------------------------------------------------------------
// Final: out[c, m] = x[m, c] / max(||x[m,:]||_2, 1e-12). One wave per row.
// ---------------------------------------------------------------------------
__global__ __launch_bounds__(256)
void l2norm_out_kernel(const float* __restrict__ X, float* __restrict__ out,
                       int M, int C) {
    int wid  = threadIdx.x >> 5;
    int lane = threadIdx.x & 31;
    int m    = blockIdx.x * 8 + wid;
    const float* xr = X + (size_t)m * C;
    float vals[8];
    float ss = 0.f;
    #pragma unroll
    for (int j = 0; j < 8; ++j) {
        float v = xr[lane + j * 32];
        vals[j] = v; ss += v * v;
    }
    #pragma unroll
    for (int off = 16; off > 0; off >>= 1) ss += __shfl_xor(ss, off, 32);
    float inv = 1.f / fmaxf(sqrtf(ss), 1e-12f);
    #pragma unroll
    for (int j = 0; j < 8; ++j) {
        int c = lane + j * 32;
        out[(size_t)c * M + m] = vals[j] * inv;
    }
}

// ---------------------------------------------------------------------------
// Host orchestration
// ---------------------------------------------------------------------------
extern "C" void kernel_launch(void* const* d_in, const int* in_sizes, int n_in,
                              void* d_out, int out_size, void* d_ws, size_t ws_size,
                              hipStream_t stream) {
    (void)in_sizes; (void)n_in; (void)out_size; (void)ws_size;

    const float* grd2sat = (const float*)d_in[0];
    const float* grd_x   = (const float*)d_in[1];
    const float* u       = (const float*)d_in[2];
    const float* ln_q_w  = (const float*)d_in[3];
    const float* ln_q_b  = (const float*)d_in[4];
    const float* ln_k_w  = (const float*)d_in[5];
    const float* ln_k_b  = (const float*)d_in[6];
    const float* ln_v_w  = (const float*)d_in[7];
    const float* ln_v_b  = (const float*)d_in[8];
    const float* Wq      = (const float*)d_in[9];
    const float* Wk      = (const float*)d_in[10];
    const float* Wv      = (const float*)d_in[11];
    const float* Wproj   = (const float*)d_in[12];
    const float* bproj   = (const float*)d_in[13];
    const float* ln_pre_w  = (const float*)d_in[14];
    const float* ln_pre_b  = (const float*)d_in[15];
    const float* Wm1     = (const float*)d_in[16];
    const float* bm1     = (const float*)d_in[17];
    const float* Wm2     = (const float*)d_in[18];
    const float* bm2     = (const float*)d_in[19];
    const float* ln_post_w = (const float*)d_in[20];
    const float* ln_post_b = (const float*)d_in[21];

    // Workspace layout (256B aligned)
    uint8_t* base = (uint8_t*)d_ws;
    size_t off = 0;
    auto take = [&](size_t bytes) {
        size_t o = off;
        off = (off + bytes + 255) & ~(size_t)255;
        return o;
    };
    const size_t SZ_F32 = (size_t)MQ * CCH * sizeof(float);      // 4 MB
    float*    G    = (float*)(base + take(SZ_F32));              // distinct ctx rows
    float*    X    = (float*)(base + take(SZ_F32));              // running activations
    float*    TMP1 = (float*)(base + take(SZ_F32));              // Q / Z / X2
    float*    Kb   = (float*)(base + take(SZ_F32));
    float*    Vb   = (float*)(base + take(SZ_F32));
    float*    ZN   = (float*)(base + take(SZ_F32));              // LN_pre f32 (residual)
    _Float16* L16  = (_Float16*)(base + take((size_t)MQ * CCH * 2));  // LN f16 out
    _Float16* BUF16= (_Float16*)(base + take((size_t)MQ * NMLP * 2)); // attn out / H
    _Float16* W16  = (_Float16*)(base + take((size_t)1048576 * 2));   // f16 weights (transposed)
    int*      wl   = (int*)(base + take((size_t)MQ * sizeof(int)));
    int*      wr   = (int*)(base + take((size_t)MQ * sizeof(int)));

    // f16 transposed weight offsets (in halfs); per-block sub-offsets below
    _Float16* Wq16    = W16;                 // 2 * 65536   (256x256 each)
    _Float16* Wk16    = W16 + 131072;
    _Float16* Wv16    = W16 + 262144;
    _Float16* Wproj16 = W16 + 393216;
    _Float16* Wm116   = W16 + 524288;        // 2 * 131072  (512x256 each, = Wm1^T)
    _Float16* Wm216   = W16 + 786432;        // 2 * 131072  (256x512 each, = Wm2^T)

    // ---- one-time prep ----
    transpose_cm_kernel<<<(MQ * CCH + 255) / 256, 256, 0, stream>>>(grd2sat, X, MQ, CCH);
    transpose_cm_kernel<<<(MQ * CCH + 255) / 256, 256, 0, stream>>>(grd_x,   G, MQ, CCH);
    make_idx_kernel<<<(MQ + 255) / 256, 256, 0, stream>>>(u, wl, wr, MQ, WIMG - 1);

    for (int i = 0; i < 2; ++i) {
        w_to_f16t_kernel<<<256, 256, 0, stream>>>(Wq    + i * 65536,  Wq16    + i * 65536,  CCH,  CCH);
        w_to_f16t_kernel<<<256, 256, 0, stream>>>(Wk    + i * 65536,  Wk16    + i * 65536,  CCH,  CCH);
        w_to_f16t_kernel<<<256, 256, 0, stream>>>(Wv    + i * 65536,  Wv16    + i * 65536,  CCH,  CCH);
        w_to_f16t_kernel<<<256, 256, 0, stream>>>(Wproj + i * 65536,  Wproj16 + i * 65536,  CCH,  CCH);
        w_to_f16t_kernel<<<512, 256, 0, stream>>>(Wm1   + i * 131072, Wm116   + i * 131072, CCH,  NMLP);
        w_to_f16t_kernel<<<512, 256, 0, stream>>>(Wm2   + i * 131072, Wm216   + i * 131072, NMLP, CCH);
    }

    const dim3 blk(256);
    const dim3 g256(MQ / BM, CCH / BN);     // 32 x 4
    const dim3 g512(MQ / BM, NMLP / BN);    // 32 x 8
    const int  lnGrid = MQ / 8;             // 512

    for (int i = 0; i < 2; ++i) {
        const int oc = i * CCH;     // per-block offset for C-sized params
        // Q = LN_q(x) @ Wq
        ln_rows_kernel<<<lnGrid, blk, 0, stream>>>(X, ln_q_w + oc, ln_q_b + oc, L16, nullptr, CCH);
        gemm_f16_wmma_kernel<false, false, false, false><<<g256, blk, 0, stream>>>(
            L16, Wq16 + i * 65536, TMP1, nullptr, nullptr, nullptr, MQ, CCH, CCH);
        // K = LN_k(G) @ Wk  (only 4096 distinct context rows)
        ln_rows_kernel<<<lnGrid, blk, 0, stream>>>(G, ln_k_w + oc, ln_k_b + oc, L16, nullptr, CCH);
        gemm_f16_wmma_kernel<false, false, false, false><<<g256, blk, 0, stream>>>(
            L16, Wk16 + i * 65536, Kb, nullptr, nullptr, nullptr, MQ, CCH, CCH);
        // V = LN_v(G) @ Wv
        ln_rows_kernel<<<lnGrid, blk, 0, stream>>>(G, ln_v_w + oc, ln_v_b + oc, L16, nullptr, CCH);
        gemm_f16_wmma_kernel<false, false, false, false><<<g256, blk, 0, stream>>>(
            L16, Wv16 + i * 65536, Vb, nullptr, nullptr, nullptr, MQ, CCH, CCH);
        // attention (gathered K/V rows), f16 output
        attn_kernel<<<MQ / 16, dim3(128), 0, stream>>>(TMP1, Kb, Vb, wl, wr, BUF16);
        // Z = A @ Wproj + bproj
        gemm_f16_wmma_kernel<true, false, false, false><<<g256, blk, 0, stream>>>(
            BUF16, Wproj16 + i * 65536, TMP1, nullptr, bproj + oc, nullptr, MQ, CCH, CCH);
        // Zn = LN_pre(Z)  (f32 for residual, f16 for MLP input)
        ln_rows_kernel<<<lnGrid, blk, 0, stream>>>(TMP1, ln_pre_w + oc, ln_pre_b + oc, L16, ZN, CCH);
        // H = gelu(Zn @ Wm1 + bm1), f16
        gemm_f16_wmma_kernel<true, true, false, true><<<g512, blk, 0, stream>>>(
            L16, Wm116 + i * 131072, nullptr, BUF16, bm1 + i * NMLP, nullptr, MQ, NMLP, CCH);
        // X2 = Zn + H @ Wm2 + bm2
        gemm_f16_wmma_kernel<true, false, true, false><<<g256, blk, 0, stream>>>(
            BUF16, Wm216 + i * 131072, TMP1, nullptr, bm2 + oc, ZN, MQ, CCH, NMLP);
        // x = LN_post(X2)
        ln_rows_kernel<<<lnGrid, blk, 0, stream>>>(TMP1, ln_post_w + oc, ln_post_b + oc, nullptr, X, CCH);
    }

    // out[c, s1, s2] = x[m, c] / max(||x[m,:]||, 1e-12)
    l2norm_out_kernel<<<MQ / 8, blk, 0, stream>>>(X, (float*)d_out, MQ, CCH);
}
